// LocalGlobalAttention_1975684956468
// MI455X (gfx1250) — matrix operand order repaired and checked by reading, not compiled, not run on hardware
//
#include <hip/hip_runtime.h>
#include <hip/hip_bf16.h>

// ---------------------------------------------------------------------------
// LocalGlobalAttention for MI455X (gfx1250, wave32):
//   - bf16 WMMA 16x16x32 for all three matmul stages (f32 accumulate)
//   - Tensor Data Mover (tensor_load_to_lds + TENSORcnt) double-buffered
//     tile staging for the projection GEMMs
// ---------------------------------------------------------------------------

typedef __bf16 bf16_t;
typedef __attribute__((ext_vector_type(16))) __bf16 v16bf;
typedef __attribute__((ext_vector_type(8)))  float  v8f;
typedef unsigned int u32x4 __attribute__((ext_vector_type(4)));
typedef int          i32x8 __attribute__((ext_vector_type(8)));
typedef int          i32x4 __attribute__((ext_vector_type(4)));

#define NTOK   2048
#define CDIM   1024
#define HEADS  16
#define DH     64
#define WIN    128
#define SSTRIDE 196   // attention LDS row: 160 local + 32 global + 4 pad (f32)
#define LDSW   40     // GEMM LDS row stride in bf16 (32 data + 4DW TDM pad)

static __device__ __forceinline__ bf16_t f2bf(float f) {
    unsigned u = __builtin_bit_cast(unsigned, f);
    unsigned r = (u + 0x7FFFu + ((u >> 16) & 1u)) >> 16;
    return __builtin_bit_cast(bf16_t, (unsigned short)r);
}

static __device__ __forceinline__ v8f wmma_bf16(v16bf a, v16bf b, v8f c) {
    return __builtin_amdgcn_wmma_f32_16x16x32_bf16(
        false, a, false, b, (short)0, c, false, false);
}

// TDM: load a [128 rows x 32 cols] bf16 tile (row stride = Kelems) from
// global into LDS at lds_off, padding each 64B row with 16B -> stride 80B.
static __device__ __forceinline__ void tdm_load_tile(
    unsigned lds_off, const bf16_t* gtile, unsigned tensor_rows, unsigned Kelems)
{
    unsigned long long ga = (unsigned long long)(size_t)gtile;
    u32x4 g0;
    g0[0] = 1u;                                   // count=1 (valid), load, user
    g0[1] = lds_off;                              // lds_addr (bytes)
    g0[2] = (unsigned)(ga & 0xFFFFFFFFu);         // global_addr[31:0]
    g0[3] = (unsigned)((ga >> 32) & 0x1FFFFFFu)   // global_addr[56:32]
          | (2u << 30);                           // type = 2 (image)
    i32x8 g1;
    g1[0] = (int)((1u << 16)                      // data_size = 2 bytes
                | (1u << 20)                      // pad_enable
                | (3u << 22)                      // pad_interval: 16 DWORDs
                | (3u << 25));                    // pad_amount: 4 DWORDs
    g1[1] = (int)((Kelems & 0xFFFFu) << 16);      // tensor_dim0[15:0]
    g1[2] = (int)(((Kelems >> 16) & 0xFFFFu)      // tensor_dim0[31:16]
                | ((tensor_rows & 0xFFFFu) << 16));// tensor_dim1[15:0]
    g1[3] = (int)(((tensor_rows >> 16) & 0xFFFFu) // tensor_dim1[31:16]
                | (32u << 16));                   // tile_dim0 = 32
    g1[4] = 128;                                  // tile_dim1 = 128 (tile_dim2=0)
    g1[5] = (int)Kelems;                          // tensor_dim0_stride[31:0]
    g1[6] = 0;
    g1[7] = 0;
    i32x4 z4 = {0, 0, 0, 0};
    i32x8 z8 = {0, 0, 0, 0, 0, 0, 0, 0};
    __builtin_amdgcn_tensor_load_to_lds(g0, g1, z4, z4, z8, 0);
}

// ---------------------------------------------------------------------------
// fp32 -> bf16 elementwise conversion (one pass; GEMMs then stream bf16)
// ---------------------------------------------------------------------------
__global__ __launch_bounds__(256) void cvt_bf16(const float* __restrict__ in,
                                                bf16_t* __restrict__ out, int n) {
    int i = (blockIdx.x * 256 + threadIdx.x) * 4;
    if (i + 3 < n) {
        #pragma unroll
        for (int t = 0; t < 4; ++t) out[i + t] = f2bf(in[i + t]);
    } else {
        for (int t = 0; t < 4 && i + t < n; ++t) out[i + t] = f2bf(in[i + t]);
    }
}

// ---------------------------------------------------------------------------
// Tiled GEMM: C[M,Ncols] = A[M,K] * W[Ncols,K]^T, bf16 in, f32 accumulate.
// 128x128 block tile, 256 threads (8 waves), each wave a 32x64 patch.
// TDM double-buffered LDS staging; wave 0 drives the DMA.
// MODE 0: scatter bf16 q/k/v [b,h,n,d].  MODE 1: fp32 out + bias.
// ---------------------------------------------------------------------------
template <int MODE>
__global__ __launch_bounds__(256) void gemm_wmma(
    const bf16_t* __restrict__ A, const bf16_t* __restrict__ W,
    const float* __restrict__ bias,
    bf16_t* __restrict__ qb, bf16_t* __restrict__ kb, bf16_t* __restrict__ vb,
    float* __restrict__ out, int K, int Mrows, int Ncols)
{
    __shared__ bf16_t As[2][128 * LDSW];
    __shared__ bf16_t Bs[2][128 * LDSW];

    const int tid  = threadIdx.x;
    const int wave = tid >> 5;
    const int lane = tid & 31;
    const int half = lane >> 4;
    const int ln   = lane & 15;
    const int wm   = wave >> 1;            // 0..3 -> M offset wm*32
    const int wn   = wave & 1;             // 0..1 -> N offset wn*64
    const int n0   = blockIdx.x * 128;
    const int m0   = blockIdx.y * 128;

    const int KT = K >> 5;                 // 32-wide K steps

    v8f acc[8];
    #pragma unroll
    for (int i = 0; i < 8; ++i) acc[i] = v8f{};

    // prologue: stage k-step 0 into buffer 0
    if (wave == 0) {
        tdm_load_tile((unsigned)(size_t)&As[0][0], A + (size_t)m0 * K, Mrows, K);
        tdm_load_tile((unsigned)(size_t)&Bs[0][0], W + (size_t)n0 * K, Ncols, K);
    }

    for (int kt = 0; kt < KT; ++kt) {
        const int buf = kt & 1;
        if (wave == 0) {
            if (kt + 1 < KT) {
                const int k1 = (kt + 1) << 5;
                tdm_load_tile((unsigned)(size_t)&As[buf ^ 1][0],
                              A + (size_t)m0 * K + k1, Mrows, K);
                tdm_load_tile((unsigned)(size_t)&Bs[buf ^ 1][0],
                              W + (size_t)n0 * K + k1, Ncols, K);
                __builtin_amdgcn_s_wait_tensorcnt(2);  // current pair done
            } else {
                __builtin_amdgcn_s_wait_tensorcnt(0);
            }
        }
        __syncthreads();   // TDM data visible to all waves

        v16bf a[2], b[4];
        #pragma unroll
        for (int s = 0; s < 2; ++s) {
            const int ra = wm * 32 + s * 16 + ln;
            #pragma unroll
            for (int e = 0; e < 16; ++e) {
                int kk = e + 16 * half;   // A-frag swizzle
                a[s][e] = As[buf][ra * LDSW + kk];
            }
        }
        #pragma unroll
        for (int s = 0; s < 4; ++s) {
            const int rb = wn * 64 + s * 16 + ln;
            #pragma unroll
            for (int e = 0; e < 16; ++e)
                b[s][e] = Bs[buf][rb * LDSW + e + 16 * half]; // B-frag swizzle
        }
        #pragma unroll
        for (int sm = 0; sm < 2; ++sm)
            #pragma unroll
            for (int sn = 0; sn < 4; ++sn)
                acc[sm * 4 + sn] = wmma_bf16(a[sm], b[sn], acc[sm * 4 + sn]);

        __syncthreads();   // all reads done before buffer is re-filled
    }

    // C/D layout: VGPR r, lane -> M = r + 8*half, N = ln
    #pragma unroll
    for (int sm = 0; sm < 2; ++sm)
    #pragma unroll
    for (int sn = 0; sn < 4; ++sn) {
        v8f c = acc[sm * 4 + sn];
        #pragma unroll
        for (int r = 0; r < 8; ++r) {
            int m  = m0 + wm * 32 + sm * 16 + r + 8 * half;
            int cg = n0 + wn * 64 + sn * 16 + ln;
            float val = c[r];
            if constexpr (MODE == 0) {
                int s    = cg >> 10;          // 0=q 1=k 2=v
                int rest = cg & 1023;
                int h    = rest >> 6;
                int d    = rest & 63;
                int btok = m >> 11;
                int n    = m & 2047;
                size_t off = (((size_t)(btok * HEADS + h)) * NTOK + n) * DH + d;
                bf16_t x = f2bf(val);
                if (s == 0)      qb[off] = x;
                else if (s == 1) kb[off] = x;
                else             vb[off] = x;
            } else {
                out[(size_t)m * CDIM + cg] = val + bias[cg];
            }
        }
    }
}

// ---------------------------------------------------------------------------
// Attention: one wave per (b, h, 16-row query tile); 4 waves / block.
// Local causal 128-window (span 144 -> 160 padded) + separate 16-key global
// softmax. WMMA scores -> LDS, fp32 softmax, bf16 P x V via WMMA.
// Output written as bf16 so the out-projection GEMM can TDM-stream it.
// ---------------------------------------------------------------------------
__global__ __launch_bounds__(128) void attn_wmma(
    const bf16_t* __restrict__ qb, const bf16_t* __restrict__ kb,
    const bf16_t* __restrict__ vb, bf16_t* __restrict__ attn_out)
{
    __shared__ float smem[4 * 16 * SSTRIDE];

    const int tid  = threadIdx.x;
    const int wave = tid >> 5;
    const int lane = tid & 31;
    const int half = lane >> 4;
    const int ln   = lane & 15;

    const int task = blockIdx.x * 4 + wave;   // 0..4095
    const int bh   = task >> 7;
    const int qt   = task & 127;
    const int i0   = qt << 4;

    const bf16_t* q0 = qb + (size_t)bh * NTOK * DH;
    const bf16_t* k0 = kb + (size_t)bh * NTOK * DH;
    const bf16_t* v0 = vb + (size_t)bh * NTOK * DH;
    float* S = &smem[wave * 16 * SSTRIDE];

    v16bf qa[2];
    #pragma unroll
    for (int ch = 0; ch < 2; ++ch)
        #pragma unroll
        for (int e = 0; e < 16; ++e) {
            int d = e + 16 * half + 32 * ch;
            qa[ch][e] = q0[(size_t)(i0 + ln) * DH + d];
        }

    for (int kt = 0; kt < 9; ++kt) {          // local score tiles
        int j  = i0 - WIN + kt * 16 + ln;
        int jc = j < 0 ? 0 : (j >= NTOK ? NTOK - 1 : j);
        v16bf bk[2];
        #pragma unroll
        for (int ch = 0; ch < 2; ++ch)
            #pragma unroll
            for (int e = 0; e < 16; ++e) {
                int d = e + 16 * half + 32 * ch;
                bk[ch][e] = k0[(size_t)jc * DH + d];
            }
        v8f c = v8f{};
        c = wmma_bf16(qa[0], bk[0], c);
        c = wmma_bf16(qa[1], bk[1], c);
        #pragma unroll
        for (int r = 0; r < 8; ++r)
            S[(r + 8 * half) * SSTRIDE + kt * 16 + ln] = c[r];
    }
    {                                          // global score tile
        int j = ln * WIN;
        v16bf bk[2];
        #pragma unroll
        for (int ch = 0; ch < 2; ++ch)
            #pragma unroll
            for (int e = 0; e < 16; ++e) {
                int d = e + 16 * half + 32 * ch;
                bk[ch][e] = k0[(size_t)j * DH + d];
            }
        v8f c = v8f{};
        c = wmma_bf16(qa[0], bk[0], c);
        c = wmma_bf16(qa[1], bk[1], c);
        #pragma unroll
        for (int r = 0; r < 8; ++r)
            S[(r + 8 * half) * SSTRIDE + 160 + ln] = c[r];
    }
    __syncthreads();

    if (lane < 16) {                           // per-row softmax, fp32
        const int row = lane;
        const int i   = i0 + row;
        float*  sf = &S[row * SSTRIDE];
        bf16_t* pb = (bf16_t*)sf;
        const float scale = 0.125f;

        float m = -3.0e38f;
        for (int c = 0; c < 160; ++c) {
            int j = i0 - WIN + c;
            bool valid = (j >= 0) && (j <= i) && (i - j <= WIN);
            if (valid) { float s = sf[c] * scale; sf[c] = s; m = fmaxf(m, s); }
            else       { sf[c] = -3.0e38f; }
        }
        float sum = 0.f;
        for (int c = 0; c < 160; ++c) {
            float s = sf[c];
            float e = (s > -1.0e37f) ? __expf(s - m) : 0.f;
            sf[c] = e; sum += e;
        }
        float inv = 1.f / sum;
        for (int c = 0; c < 160; ++c)
            pb[c] = f2bf(sf[c] * inv);

        float mg = -3.0e38f;
        for (int g = 0; g < 16; ++g) { float s = sf[160 + g] * scale; sf[160 + g] = s; mg = fmaxf(mg, s); }
        float sg = 0.f;
        for (int g = 0; g < 16; ++g) { float e = __expf(sf[160 + g] - mg); sf[160 + g] = e; sg += e; }
        float invg = 1.f / sg;
        for (int g = 0; g < 32; ++g)
            pb[160 + g] = (g < 16) ? f2bf(sf[160 + g] * invg) : f2bf(0.f);
    }
    __syncthreads();

    v8f o[4] = {v8f{}, v8f{}, v8f{}, v8f{}};
    for (int c5 = 0; c5 < 5; ++c5) {           // local P x V
        v16bf pa;
        const bf16_t* prow = (const bf16_t*)&S[ln * SSTRIDE];
        #pragma unroll
        for (int e = 0; e < 16; ++e) {
            int kc = e + 16 * half + 32 * c5;
            pa[e] = prow[kc];
        }
        #pragma unroll
        for (int dt = 0; dt < 4; ++dt) {
            v16bf bv;
            #pragma unroll
            for (int e = 0; e < 16; ++e) {
                int kk = e + 16 * half;
                int j  = i0 - WIN + c5 * 32 + kk;
                int jc = j < 0 ? 0 : (j >= NTOK ? NTOK - 1 : j);
                bv[e] = v0[(size_t)jc * DH + dt * 16 + ln];
            }
            o[dt] = wmma_bf16(pa, bv, o[dt]);
        }
    }
    {                                          // global P x V (16 + 16 pad)
        v16bf pa;
        const bf16_t* prow = (const bf16_t*)&S[ln * SSTRIDE];
        #pragma unroll
        for (int e = 0; e < 16; ++e) {
            int kc = e + 16 * half;
            pa[e] = prow[160 + kc];
        }
        #pragma unroll
        for (int dt = 0; dt < 4; ++dt) {
            v16bf bv;
            #pragma unroll
            for (int e = 0; e < 16; ++e) {
                int kk = e + 16 * half;
                int j  = (kk < 16) ? kk * WIN : 0;
                bv[e] = v0[(size_t)j * DH + dt * 16 + ln];
            }
            o[dt] = wmma_bf16(pa, bv, o[dt]);
        }
    }

    const int b = bh >> 4, h = bh & 15;
    #pragma unroll
    for (int dt = 0; dt < 4; ++dt)
        #pragma unroll
        for (int r = 0; r < 8; ++r) {
            int i = i0 + r + 8 * half;
            int d = dt * 16 + ln;
            attn_out[((size_t)(b * NTOK + i)) * CDIM + h * DH + d] = f2bf(o[dt][r]);
        }
}

// ---------------------------------------------------------------------------
extern "C" void kernel_launch(void* const* d_in, const int* in_sizes, int n_in,
                              void* d_out, int out_size, void* d_ws, size_t ws_size,
                              hipStream_t stream) {
    (void)in_sizes; (void)n_in; (void)out_size; (void)ws_size;
    const float* x     = (const float*)d_in[0];
    const float* w_qkv = (const float*)d_in[1];
    const float* w_out = (const float*)d_in[2];
    const float* b_out = (const float*)d_in[3];
    float* out = (float*)d_out;

    // workspace layout (bytes)
    const size_t SZ_QKV  = (size_t)2 * HEADS * NTOK * DH * 2;   // 8 MB each
    char* wsb = (char*)d_ws;
    bf16_t* qb     = (bf16_t*)(wsb);
    bf16_t* kb     = (bf16_t*)(wsb + SZ_QKV);
    bf16_t* vb     = (bf16_t*)(wsb + 2 * SZ_QKV);
    bf16_t* xbf    = (bf16_t*)(wsb + 3 * SZ_QKV);               // 4096x1024
    bf16_t* wqkvbf = (bf16_t*)(wsb + 3 * SZ_QKV + 8388608);     // 3072x1024
    bf16_t* woutbf = (bf16_t*)(wsb + 3 * SZ_QKV + 8388608 + 6291456); // 1024x1024
    bf16_t* attnbf = (bf16_t*)(wsb + 3 * SZ_QKV + 8388608 + 6291456 + 2097152);

    // 0) one-time fp32 -> bf16 conversions
    cvt_bf16<<<4096, 256, 0, stream>>>(x,     xbf,    4194304);
    cvt_bf16<<<3072, 256, 0, stream>>>(w_qkv, wqkvbf, 3145728);
    cvt_bf16<<<1024, 256, 0, stream>>>(w_out, woutbf, 1048576);

    // 1) QKV projection: [4096,1024] x [1024,3072]
    gemm_wmma<0><<<dim3(24, 32), 256, 0, stream>>>(
        xbf, wqkvbf, nullptr, qb, kb, vb, nullptr, CDIM, 4096, 3072);

    // 2) local + global attention (bf16 out)
    attn_wmma<<<1024, 128, 0, stream>>>(qb, kb, vb, attnbf);

    // 3) output projection: [4096,1024] x [1024,1024] + bias
    gemm_wmma<1><<<dim3(8, 32), 256, 0, stream>>>(
        attnbf, woutbf, b_out, nullptr, nullptr, nullptr, out, CDIM, 4096, 1024);
}
